// MultiScaleVGGCluster_65429531787926
// MI455X (gfx1250) — matrix operand order repaired and checked by reading, not compiled, not run on hardware
//
#include <hip/hip_runtime.h>
#include <hip/hip_bf16.h>

typedef __attribute__((ext_vector_type(16))) _Float16 v16h;
typedef __attribute__((ext_vector_type(8)))  _Float16 v8h;
typedef __attribute__((ext_vector_type(8)))  float    v8f;

// ---------------------------------------------------------------------------
// Codebook prep: f32 -> f16 copy (for WMMA A operand) + per-code ||c||^2.
// ---------------------------------------------------------------------------
template<int C>
__global__ __launch_bounds__(256)
void cb_prep_kernel(const float* __restrict__ cb,
                    _Float16* __restrict__ cb16,
                    float* __restrict__ cnorm)
{
    const int e = blockIdx.x;
    __shared__ float red[256];
    float ss = 0.0f;
    for (int k = threadIdx.x; k < C; k += 256) {
        const float v = cb[(size_t)e * C + k];
        cb16[(size_t)e * C + k] = (_Float16)v;
        ss += v * v;
    }
    red[threadIdx.x] = ss;
    __syncthreads();
    for (int s = 128; s > 0; s >>= 1) {
        if (threadIdx.x < s) red[threadIdx.x] += red[threadIdx.x + s];
        __syncthreads();
    }
    if (threadIdx.x == 0) cnorm[e] = red[0];
}

// ---------------------------------------------------------------------------
// Fused VQ: D^T = cb_tile(16xK) x zf_tile^T(Kx16) via v_wmma_f32_16x16x32_f16.
// Each wave owns NSUB 16-feature B sub-tiles held in registers; one A
// fragment (codebook) feeds NSUB independent WMMAs -> codebook traffic and
// vmem-per-wmma divided by NSUB, with independent back-to-back WMMAs for
// XDL pipelining. NSUB chosen per scale to stay inside the 1024-VGPR/wave
// budget (compiler uses s_set_vgpr_msb for >256 logical VGPRs).
// ---------------------------------------------------------------------------
template<int C, int NE, int HW, int NSUB>
__global__ __launch_bounds__(128)
void vq_argmin_kernel(const float* __restrict__ feat,
                      const _Float16* __restrict__ cb16,
                      const float* __restrict__ cnorm,
                      float* __restrict__ idx_out,
                      float* __restrict__ tile_loss)
{
    constexpr int KCH = C / 32;   // 32-wide K chunks

    __shared__ float s_cnorm[NE];
    for (int i = threadIdx.x; i < NE; i += 128) s_cnorm[i] = cnorm[i];
    __syncthreads();

    const int lane = threadIdx.x & 31;
    const int wave = threadIdx.x >> 5;
    const int tile = blockIdx.x * 4 + wave;    // (16*NSUB)-feature tile id
    const int n0   = tile * (16 * NSUB);
    const int b    = n0 / HW;                  // HW is a multiple of 16*NSUB
    const int p    = n0 % HW;

    const int nloc  = lane & 15;              // feature col within sub-tile
    const int khalf = (lane >> 4) * 16;       // B-frag K half
    const int ka    = (lane >> 4) * 8;        // A-frag K sub-offset
    const int crow  = lane & 15;              // A-frag code row

    // Load NSUB 16-feature B sub-tiles into registers; f32 ||z||^2 on the fly.
    v16h  bfrag[NSUB][KCH];
    float znorm[NSUB];
#pragma unroll
    for (int s = 0; s < NSUB; ++s) {
        const float* __restrict__ fbase =
            feat + (size_t)b * C * HW + p + s * 16 + nloc;
        float zn = 0.0f;
#pragma unroll
        for (int kk = 0; kk < KCH; ++kk) {
#pragma unroll
            for (int j = 0; j < 16; ++j) {
                const float v = fbase[(size_t)(kk * 32 + (j < 8 ? ka + j : 16 + ka + (j - 8))) * HW];
                zn += v * v;
                bfrag[s][kk][j] = (_Float16)v;
            }
        }
        znorm[s] = zn;
    }

    float minv[NSUB];
    int   minc[NSUB];
#pragma unroll
    for (int s = 0; s < NSUB; ++s) { minv[s] = 3.0e38f; minc[s] = 0; }

    const _Float16* __restrict__ arow0 = cb16 + (size_t)crow * C + ka;
    for (int ct = 0; ct < NE / 16; ++ct) {
        const int cbase = ct * 16;
        const _Float16* __restrict__ arow = arow0 + (size_t)cbase * C;
        v8f acc[NSUB];
#pragma unroll
        for (int s = 0; s < NSUB; ++s) acc[s] = (v8f){};
#pragma unroll
        for (int kk = 0; kk < KCH; ++kk) {
            const v8h lo = *(const v8h*)(arow + kk * 32);
            const v8h hi = *(const v8h*)(arow + kk * 32 + 16);
            v16h a;
#pragma unroll
            for (int i = 0; i < 8; ++i) { a[i] = lo[i]; a[8 + i] = hi[i]; }
#pragma unroll
            for (int s = 0; s < NSUB; ++s) {
                acc[s] = __builtin_amdgcn_wmma_f32_16x16x32_f16(
                             false, a, false, bfrag[s][kk],
                             (short)0, acc[s], false, false);
            }
        }
        // score = ||c||^2 - 2*dot ; running argmin over this lane's 8 codes
#pragma unroll
        for (int j = 0; j < 8; ++j) {
            const int   code = cbase + ((lane >> 4) << 3) + j;
            const float cn   = s_cnorm[code];
#pragma unroll
            for (int s = 0; s < NSUB; ++s) {
                const float d = cn - 2.0f * acc[s][j];
                if (d < minv[s]) { minv[s] = d; minc[s] = code; }
            }
        }
    }

    // Lane l and lane l^16 hold the other 8 codes per tile and the
    // complementary K-half of ||z||^2 — combine with one xor-16 shuffle.
    float lt = 0.0f;
#pragma unroll
    for (int s = 0; s < NSUB; ++s) {
        const float om = __shfl_xor(minv[s], 16, 32);
        const int   oc = __shfl_xor(minc[s], 16, 32);
        if (om < minv[s] || (om == minv[s] && oc < minc[s])) {
            minv[s] = om; minc[s] = oc;
        }
        const float zn = znorm[s] + __shfl_xor(znorm[s], 16, 32);
        lt += zn + minv[s];
        if (lane < 16) idx_out[n0 + s * 16 + lane] = (float)minc[s];
    }

    // Sum the per-feature loss terms across the 16 feature columns.
#pragma unroll
    for (int w = 1; w < 16; w <<= 1) lt += __shfl_xor(lt, w, 32);
    if (lane == 0) tile_loss[tile] = lt;   // deterministic reduce later
}

// ---------------------------------------------------------------------------
// Deterministic final reduction of per-tile loss sums -> scalar total loss.
// ---------------------------------------------------------------------------
__global__ __launch_bounds__(256)
void finalize_kernel(const float* __restrict__ ts1, int n1,
                     const float* __restrict__ ts2, int n2,
                     const float* __restrict__ ts3, int n3,
                     float* __restrict__ out,
                     float inv1, float inv2, float inv3)
{
    __shared__ float red[256];
    float a1 = 0.f, a2 = 0.f, a3 = 0.f;
    for (int i = threadIdx.x; i < n1; i += 256) a1 += ts1[i];
    for (int i = threadIdx.x; i < n2; i += 256) a2 += ts2[i];
    for (int i = threadIdx.x; i < n3; i += 256) a3 += ts3[i];
    red[threadIdx.x] = a1 * inv1 + a2 * inv2 + a3 * inv3;
    __syncthreads();
    for (int s = 128; s > 0; s >>= 1) {
        if (threadIdx.x < s) red[threadIdx.x] += red[threadIdx.x + s];
        __syncthreads();
    }
    if (threadIdx.x == 0) out[0] = red[0];
}

// ---------------------------------------------------------------------------
// Launch
// ---------------------------------------------------------------------------
extern "C" void kernel_launch(void* const* d_in, const int* in_sizes, int n_in,
                              void* d_out, int out_size, void* d_ws, size_t ws_size,
                              hipStream_t stream)
{
    // setup_inputs order: input, feat1, feat2, feat3, cb1, cb2, cb3
    const float* feat1 = (const float*)d_in[1];  // [16,256,64,64]
    const float* feat2 = (const float*)d_in[2];  // [16,512,32,32]
    const float* feat3 = (const float*)d_in[3];  // [16,512,16,16]
    const float* cb1   = (const float*)d_in[4];  // [512,256]
    const float* cb2   = (const float*)d_in[5];  // [1024,512]
    const float* cb3   = (const float*)d_in[6];  // [1024,512]
    float* out = (float*)d_out;                  // [1 + 65536 + 16384 + 4096]

    // Workspace carve-out (256B aligned slices)
    char* ws = (char*)d_ws;
    auto carve = [&](size_t bytes) -> char* {
        char* pp = ws;
        ws += (bytes + 255) & ~(size_t)255;
        return pp;
    };
    _Float16* cb16_1 = (_Float16*)carve((size_t)512  * 256 * 2);
    _Float16* cb16_2 = (_Float16*)carve((size_t)1024 * 512 * 2);
    _Float16* cb16_3 = (_Float16*)carve((size_t)1024 * 512 * 2);
    float* cnorm1 = (float*)carve(512  * 4);
    float* cnorm2 = (float*)carve(1024 * 4);
    float* cnorm3 = (float*)carve(1024 * 4);
    float* ts1 = (float*)carve(1024 * 4);   // 65536/64 tiles (NSUB=4)
    float* ts2 = (float*)carve(512  * 4);   // 16384/32 tiles (NSUB=2)
    float* ts3 = (float*)carve(128  * 4);   // 4096/32  tiles (NSUB=2)

    // 1) codebook prep
    cb_prep_kernel<256><<<512,  256, 0, stream>>>(cb1, cb16_1, cnorm1);
    cb_prep_kernel<512><<<1024, 256, 0, stream>>>(cb2, cb16_2, cnorm2);
    cb_prep_kernel<512><<<1024, 256, 0, stream>>>(cb3, cb16_3, cnorm3);

    // 2) fused WMMA VQ (4 waves/block; 16*NSUB features per wave)
    float* i1 = out + 1;
    float* i2 = i1 + 65536;
    float* i3 = i2 + 16384;
    vq_argmin_kernel<256, 512,  4096, 4><<<256, 128, 0, stream>>>(feat1, cb16_1, cnorm1, i1, ts1);
    vq_argmin_kernel<512, 1024, 1024, 2><<<128, 128, 0, stream>>>(feat2, cb16_2, cnorm2, i2, ts2);
    vq_argmin_kernel<512, 1024, 256,  2><<<32,  128, 0, stream>>>(feat3, cb16_3, cnorm3, i3, ts3);

    // 3) total loss = sum_i mean_i
    finalize_kernel<<<1, 256, 0, stream>>>(
        ts1, 1024, ts2, 512, ts3, 128, out,
        1.0f / (65536.0f * 256.0f),
        1.0f / (16384.0f * 512.0f),
        1.0f / (4096.0f  * 512.0f));
}
